// Model_57982058496062
// MI455X (gfx1250) — matrix operand order repaired and checked by reading, not compiled, not run on hardware
//
#include <hip/hip_runtime.h>
#include <hip/hip_bf16.h>

#define TOK    8192
#define EDIM   2048
#define INTER  4096
#define NE     8
#define TOPK   2

#define BM 128
#define BN 64
#define KB 32
#define AST 20   // LDS row stride in dwords: 16 data + 4 pad (bank spread)
#define NSTEP1 (EDIM / KB)    // 64
#define NSTEP2 (INTER / KB)   // 128

typedef __attribute__((ext_vector_type(8)))  float  v8f;
typedef __attribute__((ext_vector_type(16))) __bf16 bf16x16;
typedef __attribute__((ext_vector_type(2)))  __bf16 bf16x2;
typedef int v4i __attribute__((__vector_size__(4 * sizeof(int))));

#if defined(__has_builtin)
#  if __has_builtin(__builtin_amdgcn_global_load_async_to_lds_b128)
#    define USE_ASYNC 1
#  endif
#  if __has_builtin(__builtin_amdgcn_cvt_pk_bf16_f32)
#    define USE_PK_CVT 1
#  endif
#endif
#ifndef USE_ASYNC
#  define USE_ASYNC 0
#endif
#ifndef USE_PK_CVT
#  define USE_PK_CVT 0
#endif

#define TO_GLB_V4I(p) ((__attribute__((address_space(1))) v4i*)(p))
#define TO_LDS_V4I(p) ((__attribute__((address_space(3))) v4i*)(p))

__device__ __forceinline__ unsigned short f2bf(float f) {
  union { float f; unsigned int u; } v; v.f = f;
  unsigned int u = v.u;
  return (unsigned short)((u + 0x7FFFu + ((u >> 16) & 1u)) >> 16);
}
__device__ __forceinline__ unsigned int pk2(float a, float b) {
#if USE_PK_CVT
  union { bf16x2 v; unsigned int u; } r;
  r.v = __builtin_amdgcn_cvt_pk_bf16_f32(a, b);   // v_cvt_pk_bf16_f32
  return r.u;
#else
  return (unsigned int)f2bf(a) | ((unsigned int)f2bf(b) << 16);
#endif
}

// stage 32 bytes (one A-row half) global bf16 -> LDS
__device__ __forceinline__ void stage_a(const unsigned short* g, unsigned int* l, bool ok) {
#if USE_ASYNC
  (void)ok;  // garbage rows are masked at the epilogue; addresses are clamped in-bounds
  __builtin_amdgcn_global_load_async_to_lds_b128(TO_GLB_V4I(g), TO_LDS_V4I(l), 0, 0);
  __builtin_amdgcn_global_load_async_to_lds_b128(TO_GLB_V4I(g), TO_LDS_V4I(l), 16, 0);
#else
  uint4 p0 = {0,0,0,0}, p1 = {0,0,0,0};
  if (ok) { p0 = ((const uint4*)g)[0]; p1 = ((const uint4*)g)[1]; }
  ((uint4*)l)[0] = p0; ((uint4*)l)[1] = p1;
#endif
}

__device__ __forceinline__ void wait_async0() {
#if USE_ASYNC
#  if __has_builtin(__builtin_amdgcn_s_wait_asynccnt)
  __builtin_amdgcn_s_wait_asynccnt(0);
#  else
  asm volatile("s_wait_asynccnt 0x0" ::: "memory");
#  endif
#endif
}

__device__ __forceinline__ void store_w(unsigned int* d, float4 a, float4 b) {
  d[0] = pk2(a.x, a.y); d[1] = pk2(a.z, a.w);
  d[2] = pk2(b.x, b.y); d[3] = pk2(b.z, b.w);
}

union frag_t { bf16x16 v; uint4 q[2]; };

// ---------------- zero output ----------------
__global__ void zero_kernel(float* p, int n) {
  int i = blockIdx.x * blockDim.x + threadIdx.x;
  for (; i < n; i += gridDim.x * blockDim.x) p[i] = 0.0f;
}

// ---------------- fp32 x -> bf16 xb ----------------
__global__ void cvt_x_kernel(const float* __restrict__ x, unsigned int* __restrict__ xb32) {
  const int n = TOK * EDIM / 2;
  int i = blockIdx.x * blockDim.x + threadIdx.x;
  for (; i < n; i += gridDim.x * blockDim.x) {
    float2 f = ((const float2*)x)[i];
    xb32[i] = pk2(f.x, f.y);
  }
}

// ---------------- gating: softmax + top-2 renorm -> cw[T,NE] ----------------
__global__ __launch_bounds__(256) void gate_kernel(const float* __restrict__ x,
                                                   const float* __restrict__ gw,
                                                   float* __restrict__ cw) {
  const int wave = threadIdx.x >> 5;
  const int lane = threadIdx.x & 31;
  const int t = blockIdx.x * 8 + wave;

  float acc[NE];
#pragma unroll
  for (int e = 0; e < NE; e++) acc[e] = 0.0f;

  for (int c = lane; c < EDIM; c += 32) {
    float xv = x[(size_t)t * EDIM + c];
#pragma unroll
    for (int e = 0; e < NE; e++) acc[e] += xv * gw[e * EDIM + c];
  }
#pragma unroll
  for (int off = 16; off > 0; off >>= 1) {
#pragma unroll
    for (int e = 0; e < NE; e++) acc[e] += __shfl_xor(acc[e], off);
  }
  if (lane == 0) {
    float m = acc[0];
#pragma unroll
    for (int e = 1; e < NE; e++) m = fmaxf(m, acc[e]);
    float p[NE], s = 0.0f;
#pragma unroll
    for (int e = 0; e < NE; e++) { p[e] = __expf(acc[e] - m); s += p[e]; }
    float inv = 1.0f / s;
#pragma unroll
    for (int e = 0; e < NE; e++) p[e] *= inv;
    int i1 = 0; float p1 = p[0];
#pragma unroll
    for (int e = 1; e < NE; e++) if (p[e] > p1) { p1 = p[e]; i1 = e; }
    int i2 = -1; float p2 = -1.0f;
#pragma unroll
    for (int e = 0; e < NE; e++) if (e != i1 && p[e] > p2) { p2 = p[e]; i2 = e; }
    float rs = 1.0f / (p1 + p2);
#pragma unroll
    for (int e = 0; e < NE; e++) cw[(size_t)t * NE + e] = 0.0f;
    cw[(size_t)t * NE + i1] = p1 * rs;
    cw[(size_t)t * NE + i2] = p2 * rs;
  }
}

// ---------------- deterministic per-expert compaction (1 wave / expert) ----------------
__global__ void compact_kernel(const float* __restrict__ cw, int* __restrict__ lists,
                               int* __restrict__ counts) {
  const int e = blockIdx.x;
  const int lane = threadIdx.x;
  int cnt = 0;
  for (int base = 0; base < TOK; base += 32) {
    int t = base + lane;
    bool f = cw[(size_t)t * NE + e] > 0.0f;
    unsigned int mm = (unsigned int)__ballot(f);
    int pos = __popc(mm & ((1u << lane) - 1u));
    if (f) lists[e * TOK + cnt + pos] = t;
    cnt += __popc(mm);
  }
  if (lane == 0) counts[e] = cnt;
}

__global__ void scan_kernel(const int* __restrict__ counts, int* __restrict__ offsets) {
  if (threadIdx.x == 0) {
    int s = 0;
    for (int e = 0; e < NE; e++) { offsets[e] = s; s += counts[e]; }
    offsets[NE] = s;
  }
}

// ---------------- GEMM1: h = silu(X@Wg^T) * (X@Wu^T), double-buffered bf16 WMMA -------
__global__ __launch_bounds__(256) void gemm1_kernel(
    const unsigned short* __restrict__ xb, const float* __restrict__ w_gate,
    const float* __restrict__ w_up, const int* __restrict__ lists,
    const int* __restrict__ counts, const int* __restrict__ offsets,
    unsigned short* __restrict__ hb) {
  const int e = blockIdx.z;
  const int count = counts[e];
  const int mbase = blockIdx.y * BM;
  if (mbase >= count) return;
  const int nbase = blockIdx.x * BN;
  int valid = count - mbase; if (valid > BM) valid = BM;

  __shared__ unsigned int lds_a[2 * BM * AST];
  __shared__ unsigned int lds_g[2 * BN * AST];
  __shared__ unsigned int lds_u[2 * BN * AST];

  const int tid  = threadIdx.x;
  const int lane = tid & 31;
  const int wave = tid >> 5;
  const int hi   = (lane >> 4) & 1;
  const int l15  = lane & 15;

  // A staging: 2 threads per row, each covers 16 bf16 (32 bytes)
  const int arow = tid >> 1;
  const int akh  = tid & 1;
  const bool arow_ok = (arow < valid);
  const int tokenA = arow_ok ? lists[e * TOK + mbase + arow] : lists[e * TOK + mbase];

  // W staging: 4 threads per row, each covers 8 fp32
  const int wr = tid >> 2;
  const int wk = (tid & 3) * 8;

  const float* wg = w_gate + (size_t)e * INTER * EDIM;
  const float* wu = w_up   + (size_t)e * INTER * EDIM;

  const unsigned short* aglb = xb + (size_t)tokenA * EDIM + akh * 16;
  const float* gsrc = wg + (size_t)(nbase + wr) * EDIM + wk;
  const float* usrc = wu + (size_t)(nbase + wr) * EDIM + wk;
  unsigned int* const adst = &lds_a[arow * AST + akh * 8];
  unsigned int* const gdst = &lds_g[wr * AST + wk / 2];
  unsigned int* const udst = &lds_u[wr * AST + wk / 2];

  v8f accg[4], accu[4];
#pragma unroll
  for (int i = 0; i < 4; i++) {
    accg[i] = (v8f){0.f,0.f,0.f,0.f,0.f,0.f,0.f,0.f};
    accu[i] = (v8f){0.f,0.f,0.f,0.f,0.f,0.f,0.f,0.f};
  }

  // prologue: stage K-tile 0 into buffer 0
  stage_a(aglb, adst, arow_ok);
  float4 g0 = ((const float4*)gsrc)[0], g1 = ((const float4*)gsrc)[1];
  float4 u0 = ((const float4*)usrc)[0], u1 = ((const float4*)usrc)[1];
  store_w(gdst, g0, g1);
  store_w(udst, u0, u1);
  wait_async0();
  __syncthreads();

  for (int ks = 0; ks < NSTEP1; ks++) {
    const int cur = ks & 1, nxt = cur ^ 1;
    const bool more = (ks + 1) < NSTEP1;
    const int kn = (ks + 1) * KB;
    if (more) {
      stage_a(aglb + kn, adst + nxt * (BM * AST), arow_ok);   // async: lands in LDS directly
      __builtin_prefetch((const void*)(gsrc + kn + KB), 0, 1);
      __builtin_prefetch((const void*)(usrc + kn + KB), 0, 1);
      g0 = ((const float4*)(gsrc + kn))[0]; g1 = ((const float4*)(gsrc + kn))[1];
      u0 = ((const float4*)(usrc + kn))[0]; u1 = ((const float4*)(usrc + kn))[1];
    }
    const unsigned int* la = &lds_a[cur * BM * AST];
    const unsigned int* lg = &lds_g[cur * BN * AST];
    const unsigned int* lu = &lds_u[cur * BN * AST];

    // load ALL fragments first (distinct registers) so WMMAs are not
    // serialized behind a single recycled fragment register set
    frag_t fa;
    {
      const int base = (wave * 16 + l15) * AST + hi * 4;
      fa.q[0] = *(const uint4*)&la[base];
      fa.q[1] = *(const uint4*)&la[base + 8];
    }
    frag_t fg[4], fu[4];
#pragma unroll
    for (int nt = 0; nt < 4; nt++) {
      const int bb = (nt * 16 + l15) * AST + hi * 4;
      fg[nt].q[0] = *(const uint4*)&lg[bb]; fg[nt].q[1] = *(const uint4*)&lg[bb + 8];
      fu[nt].q[0] = *(const uint4*)&lu[bb]; fu[nt].q[1] = *(const uint4*)&lu[bb + 8];
    }
#pragma unroll
    for (int nt = 0; nt < 4; nt++) {
      accg[nt] = __builtin_amdgcn_wmma_f32_16x16x32_bf16(false, fa.v, false, fg[nt].v,
                                                         (short)0, accg[nt], false, false);
      accu[nt] = __builtin_amdgcn_wmma_f32_16x16x32_bf16(false, fa.v, false, fu[nt].v,
                                                         (short)0, accu[nt], false, false);
    }
    if (more) {
      store_w(gdst + nxt * (BN * AST), g0, g1);
      store_w(udst + nxt * (BN * AST), u0, u1);
      wait_async0();
    }
    __syncthreads();
  }

  // epilogue: h = silu(g) * u, bf16 store
  const int hrow0 = offsets[e] + mbase;
#pragma unroll
  for (int nt = 0; nt < 4; nt++) {
#pragma unroll
    for (int i = 0; i < 8; i++) {
      const int m = wave * 16 + hi * 8 + i;
      if (m < valid) {
        float g = accg[nt][i];
        float u = accu[nt][i];
        float h = (g / (1.0f + __expf(-g))) * u;
        hb[(size_t)(hrow0 + m) * INTER + nbase + nt * 16 + l15] = f2bf(h);
      }
    }
  }
}

// ---------------- GEMM2: out += cw * (h @ Wd^T), double-buffered, fp32 atomics --------
__global__ __launch_bounds__(256) void gemm2_kernel(
    const unsigned short* __restrict__ hb, const float* __restrict__ w_down,
    const int* __restrict__ lists, const int* __restrict__ counts,
    const int* __restrict__ offsets, const float* __restrict__ cw,
    float* __restrict__ out) {
  const int e = blockIdx.z;
  const int count = counts[e];
  const int mbase = blockIdx.y * BM;
  if (mbase >= count) return;
  const int nbase = blockIdx.x * BN;
  int valid = count - mbase; if (valid > BM) valid = BM;
  const int hrow0 = offsets[e] + mbase;

  __shared__ unsigned int lds_a[2 * BM * AST];
  __shared__ unsigned int lds_b[2 * BN * AST];

  const int tid  = threadIdx.x;
  const int lane = tid & 31;
  const int wave = tid >> 5;
  const int hi   = (lane >> 4) & 1;
  const int l15  = lane & 15;

  const int arow = tid >> 1;
  const int akh  = tid & 1;
  const bool arow_ok = (arow < valid);
  const int srcrow = arow_ok ? (hrow0 + arow) : hrow0;   // clamp in-bounds

  const int wr = tid >> 2;
  const int wk = (tid & 3) * 8;
  const float* wd = w_down + (size_t)e * EDIM * INTER;

  const unsigned short* aglb = hb + (size_t)srcrow * INTER + akh * 16;
  const float* bsrc = wd + (size_t)(nbase + wr) * INTER + wk;
  unsigned int* const adst = &lds_a[arow * AST + akh * 8];
  unsigned int* const bdst = &lds_b[wr * AST + wk / 2];

  v8f acc[4];
#pragma unroll
  for (int i = 0; i < 4; i++) acc[i] = (v8f){0.f,0.f,0.f,0.f,0.f,0.f,0.f,0.f};

  stage_a(aglb, adst, arow_ok);
  float4 b0 = ((const float4*)bsrc)[0], b1 = ((const float4*)bsrc)[1];
  store_w(bdst, b0, b1);
  wait_async0();
  __syncthreads();

  for (int ks = 0; ks < NSTEP2; ks++) {
    const int cur = ks & 1, nxt = cur ^ 1;
    const bool more = (ks + 1) < NSTEP2;
    const int kn = (ks + 1) * KB;
    if (more) {
      stage_a(aglb + kn, adst + nxt * (BM * AST), arow_ok);
      __builtin_prefetch((const void*)(bsrc + kn + KB), 0, 1);
      b0 = ((const float4*)(bsrc + kn))[0]; b1 = ((const float4*)(bsrc + kn))[1];
    }
    const unsigned int* la = &lds_a[cur * BM * AST];
    const unsigned int* lb = &lds_b[cur * BN * AST];

    frag_t fa;
    {
      const int base = (wave * 16 + l15) * AST + hi * 4;
      fa.q[0] = *(const uint4*)&la[base];
      fa.q[1] = *(const uint4*)&la[base + 8];
    }
    frag_t fb[4];
#pragma unroll
    for (int nt = 0; nt < 4; nt++) {
      const int bb = (nt * 16 + l15) * AST + hi * 4;
      fb[nt].q[0] = *(const uint4*)&lb[bb]; fb[nt].q[1] = *(const uint4*)&lb[bb + 8];
    }
#pragma unroll
    for (int nt = 0; nt < 4; nt++) {
      acc[nt] = __builtin_amdgcn_wmma_f32_16x16x32_bf16(false, fa.v, false, fb[nt].v,
                                                        (short)0, acc[nt], false, false);
    }
    if (more) {
      store_w(bdst + nxt * (BN * AST), b0, b1);
      wait_async0();
    }
    __syncthreads();
  }

  int   tok[8];
  float scl[8];
#pragma unroll
  for (int i = 0; i < 8; i++) {
    const int m = wave * 16 + hi * 8 + i;
    if (m < valid) {
      tok[i] = lists[e * TOK + mbase + m];
      scl[i] = cw[(size_t)tok[i] * NE + e];
    } else { tok[i] = 0; scl[i] = 0.0f; }
  }
#pragma unroll
  for (int nt = 0; nt < 4; nt++) {
#pragma unroll
    for (int i = 0; i < 8; i++) {
      const int m = wave * 16 + hi * 8 + i;
      if (m < valid) {
        unsafeAtomicAdd(&out[(size_t)tok[i] * EDIM + nbase + nt * 16 + l15],
                        scl[i] * acc[nt][i]);
      }
    }
  }
}

// ---------------- launch ----------------
extern "C" void kernel_launch(void* const* d_in, const int* in_sizes, int n_in,
                              void* d_out, int out_size, void* d_ws, size_t ws_size,
                              hipStream_t stream) {
  (void)in_sizes; (void)n_in; (void)ws_size;
  const float* x      = (const float*)d_in[0];
  const float* gate_w = (const float*)d_in[1];
  const float* w_gate = (const float*)d_in[2];
  const float* w_up   = (const float*)d_in[3];
  const float* w_down = (const float*)d_in[4];
  float* out = (float*)d_out;

  char* p = (char*)d_ws;
  auto alignup = [](char* q) { return (char*)(((uintptr_t)q + 255) & ~(uintptr_t)255); };
  float* cw = (float*)p;            p += (size_t)TOK * NE * 4;      p = alignup(p);
  int* counts = (int*)p;            p += 256;
  int* offsets = (int*)p;           p += 256;                       p = alignup(p);
  int* lists = (int*)p;             p += (size_t)NE * TOK * 4;      p = alignup(p);
  unsigned short* xb = (unsigned short*)p; p += (size_t)TOK * EDIM * 2;          p = alignup(p);
  unsigned short* hb = (unsigned short*)p; p += (size_t)TOK * TOPK * INTER * 2;  // ~168 MB total

  zero_kernel<<<2048, 256, 0, stream>>>(out, out_size);
  cvt_x_kernel<<<2048, 256, 0, stream>>>(x, (unsigned int*)xb);
  gate_kernel<<<TOK / 8, 256, 0, stream>>>(x, gate_w, cw);
  compact_kernel<<<NE, 32, 0, stream>>>(cw, lists, counts);
  scan_kernel<<<1, 32, 0, stream>>>(counts, offsets);
  gemm1_kernel<<<dim3(INTER / BN, TOK / BM, NE), 256, 0, stream>>>(
      xb, w_gate, w_up, lists, counts, offsets, hb);
  gemm2_kernel<<<dim3(EDIM / BN, TOK / BM, NE), 256, 0, stream>>>(
      hb, w_down, lists, counts, offsets, cw, out);
}
